// minGRU_74354473828452
// MI455X (gfx1250) — matrix operand order repaired and checked by reading, not compiled, not run on hardware
//
#include <hip/hip_runtime.h>

// ---------- types ----------
typedef __attribute__((ext_vector_type(16))) __bf16 v16bf;
typedef __attribute__((ext_vector_type(8)))  float  v8f;

#define B_SZ   4
#define S_SZ   4096
#define DIM_   1024
#define DINNER 2048
#define M_ALL  (B_SZ * S_SZ)      // 16384
#define NCHUNK 8
#define CHLEN  (S_SZ / NCHUNK)    // 512

// ---------- helpers ----------
__device__ __forceinline__ unsigned short f32_to_bf16(float f) {
    unsigned int u = __float_as_uint(f);
    u += 0x7FFFu + ((u >> 16) & 1u);          // round-to-nearest-even
    return (unsigned short)(u >> 16);
}

__device__ __forceinline__ float softplusf(float x) {
    return fmaxf(x, 0.f) + log1pf(__expf(-fabsf(x)));
}
__device__ __forceinline__ float log_g_f(float x) {
    return (x >= 0.f) ? __logf(x + 0.5f) : -softplusf(-x);
}
__device__ __forceinline__ float logaddexpf_(float a, float b) {
    float m = fmaxf(a, b);
    if (m == -__builtin_inff()) return -__builtin_inff();
    return m + log1pf(__expf(fminf(a, b) - m));
}

__device__ __forceinline__ v8f zero8() {
    v8f z = {0.f, 0.f, 0.f, 0.f, 0.f, 0.f, 0.f, 0.f};
    return z;
}

__device__ __forceinline__ void async_load_b128(unsigned lds_off,
                                                const void* gaddr) {
    // GLOBAL_LOAD_ASYNC_TO_LDS_B128 (GV mode): LDS addr from VDST VGPR,
    // 64-bit global addr from VADDR pair. Tracked with ASYNCcnt.
    asm volatile("global_load_async_to_lds_b128 %0, %1, off"
                 :: "v"(lds_off), "v"(gaddr) : "memory");
}
__device__ __forceinline__ void wait_asynccnt0() {
#if __has_builtin(__builtin_amdgcn_s_wait_asynccnt)
    __builtin_amdgcn_s_wait_asynccnt(0);
#else
    asm volatile("s_wait_asynccnt 0x0" ::: "memory");
#endif
}

// A fragment (16-bit A 16x32): lane = half*16+r holds row r, K in
// [kb..kb+7] and [kb+16..kb+23], kb = 8*half. LDS tile row-major [rows][32].
__device__ __forceinline__ v16bf load_fragA(const unsigned short* base, int row, int kb) {
    union { v16bf v; uint4 q[2]; } u;
    u.q[0] = *reinterpret_cast<const uint4*>(base + row * 32 + kb);
    u.q[1] = *reinterpret_cast<const uint4*>(base + row * 32 + kb + 16);
    return u.v;
}
// B fragment (16-bit B 32x16): lane = half*16+c holds col c, K in
// [kb..kb+7] and [kb+16..kb+23], kb = 8*half. LDS tile stored transposed: [cols][32].
__device__ __forceinline__ v16bf load_fragB(const unsigned short* baseT, int col, int kb16) {
    union { v16bf v; uint4 q[2]; } u;
    u.q[0] = *reinterpret_cast<const uint4*>(baseT + col * 32 + kb16);
    u.q[1] = *reinterpret_cast<const uint4*>(baseT + col * 32 + kb16 + 16);
    return u.v;
}

// ---------- f32 -> bf16 convert ----------
__global__ void cvt_bf16_kernel(const float* __restrict__ in,
                                unsigned short* __restrict__ out, int n) {
    for (int i = blockIdx.x * blockDim.x + threadIdx.x; i < n;
         i += gridDim.x * blockDim.x)
        out[i] = f32_to_bf16(in[i]);
}

// ---------- GEMM1 + activation: LC/LV = f(x @ W_hg) ----------
// block tile: M=128, N=64 (hidden cols) + matching 64 gate cols (n+2048).
// Double-buffered LDS; A tiles via async global->LDS DMA, B tiles via
// VGPR staging (needs transpose-on-store), issued one K-step ahead.
__global__ __launch_bounds__(256) void gemm1_act_kernel(
    const unsigned short* __restrict__ Xb,    // [16384][1024] bf16
    const unsigned short* __restrict__ Wb,    // [1024][4096]  bf16
    float* __restrict__ LC, float* __restrict__ LV) {
    __shared__ uint4 smemq[2048];                        // 32 KB
    unsigned short* SA = reinterpret_cast<unsigned short*>(smemq);  // 2 x 128x32
    unsigned short* SB = SA + 2 * 4096;                             // 2 x 128x32 (T)

    const int t    = threadIdx.x;
    const int lane = t & 31;
    const int wave = t >> 5;
    const int wm   = wave >> 1;        // 0..3
    const int wn   = wave & 1;         // 0..1
    const int rowBase = blockIdx.y * 128;
    const int n0      = blockIdx.x * 64;

    v8f acc[2][2], accg[2][2];
#pragma unroll
    for (int i = 0; i < 2; ++i)
#pragma unroll
        for (int j = 0; j < 2; ++j) { acc[i][j] = zero8(); accg[i][j] = zero8(); }

    const int arow = wm * 32 + (lane & 15);
    const int kbA  = (lane >> 4) * 8;
    const int bcol = wn * 32 + (lane & 15);
    const int kbB  = (lane >> 4) * 8;

    // B staging thread mapping: kk = k-row, c = 16-col chunk (c<4 hidden, c>=4 gate)
    const int kkB  = t >> 3;           // 0..31
    const int cB   = t & 7;            // 0..7
    const int gcolB = (cB < 4) ? (n0 + cB * 16) : (DINNER + n0 + (cB - 4) * 16);

    union BStage { uint4 q[2]; unsigned short s[16]; };

    auto issueA = [&](int kt, int buf) {
        const int k0 = kt * 32;
        unsigned short* dst = SA + buf * 4096;
#pragma unroll
        for (int it = 0; it < 2; ++it) {
            int lin = t + it * 256;            // 0..511
            int r = lin >> 2, c = lin & 3;
            const unsigned short* src = Xb + (size_t)(rowBase + r) * DIM_ + k0 + c * 8;
            unsigned loff = (unsigned)(uintptr_t)(dst + r * 32 + c * 8);
            async_load_b128(loff, src);
        }
    };
    auto loadBregs = [&](int kt, BStage& u) {
        const unsigned short* src = Wb + (size_t)(kt * 32 + kkB) * (2 * DINNER) + gcolB;
        u.q[0] = *reinterpret_cast<const uint4*>(src);
        u.q[1] = *reinterpret_cast<const uint4*>(src + 8);
    };
    auto storeB = [&](int buf, const BStage& u) {
        unsigned short* dst = SB + buf * 4096;
        int lcol = cB * 16;
#pragma unroll
        for (int i = 0; i < 16; ++i) dst[(lcol + i) * 32 + kkB] = u.s[i];
    };

    BStage bs;
    issueA(0, 0);
    loadBregs(0, bs);

    for (int kt = 0; kt < DIM_ / 32; ++kt) {
        const int buf = kt & 1;
        storeB(buf, bs);
        wait_asynccnt0();                 // this wave's async A(kt) landed in LDS
        __syncthreads();                  // s_wait_dscnt 0 + barrier
        if (kt + 1 < DIM_ / 32) {
            issueA(kt + 1, buf ^ 1);      // DMA next A tile under the WMMAs
            loadBregs(kt + 1, bs);        // start next B loads; waited at next storeB
        }
        const unsigned short* As  = SA + buf * 4096;
        const unsigned short* Bst = SB + buf * 4096;

        v16bf a0 = load_fragA(As, arow,      kbA);
        v16bf a1 = load_fragA(As, arow + 16, kbA);
        v16bf bh0 = load_fragB(Bst, bcol,           kbB);
        v16bf bh1 = load_fragB(Bst, bcol + 16,      kbB);
        v16bf bg0 = load_fragB(Bst, 64 + bcol,      kbB);
        v16bf bg1 = load_fragB(Bst, 64 + bcol + 16, kbB);

        acc[0][0]  = __builtin_amdgcn_wmma_f32_16x16x32_bf16(false, a0, false, bh0, (short)0, acc[0][0],  false, false);
        acc[0][1]  = __builtin_amdgcn_wmma_f32_16x16x32_bf16(false, a0, false, bh1, (short)0, acc[0][1],  false, false);
        acc[1][0]  = __builtin_amdgcn_wmma_f32_16x16x32_bf16(false, a1, false, bh0, (short)0, acc[1][0],  false, false);
        acc[1][1]  = __builtin_amdgcn_wmma_f32_16x16x32_bf16(false, a1, false, bh1, (short)0, acc[1][1],  false, false);
        accg[0][0] = __builtin_amdgcn_wmma_f32_16x16x32_bf16(false, a0, false, bg0, (short)0, accg[0][0], false, false);
        accg[0][1] = __builtin_amdgcn_wmma_f32_16x16x32_bf16(false, a0, false, bg1, (short)0, accg[0][1], false, false);
        accg[1][0] = __builtin_amdgcn_wmma_f32_16x16x32_bf16(false, a1, false, bg0, (short)0, accg[1][0], false, false);
        accg[1][1] = __builtin_amdgcn_wmma_f32_16x16x32_bf16(false, a1, false, bg1, (short)0, accg[1][1], false, false);
    }

    // ---- fused activation epilogue ----
#pragma unroll
    for (int fm = 0; fm < 2; ++fm)
#pragma unroll
        for (int fn = 0; fn < 2; ++fn)
#pragma unroll
            for (int i = 0; i < 8; ++i) {
                int row = rowBase + wm * 32 + fm * 16 + i + ((lane >> 4) << 3);
                int col = n0 + wn * 32 + fn * 16 + (lane & 15);
                float h = acc[fm][fn][i];
                float g = accg[fm][fn][i];
                float lc = -softplusf(g);                      // log(1-z)
                float lv = -softplusf(-g) + log_g_f(h);        // log z + log g(h)
                size_t idx = (size_t)row * DINNER + col;
                LC[idx] = lc;
                LV[idx] = lv;
            }
}

// ---------- scan phase A: per-chunk (A = sum lc, B = chunk result from -inf) ----------
__global__ void scan_reduce_kernel(const float* __restrict__ LC,
                                   const float* __restrict__ LV,
                                   float* __restrict__ CA, float* __restrict__ CB) {
    int tid = blockIdx.x * blockDim.x + threadIdx.x;   // 65536 = b*8chunks*2048e
    int e  = tid & (DINNER - 1);
    int bj = tid >> 11;                                // b*8 + j
    int b = bj >> 3, j = bj & 7;
    size_t base = ((size_t)b * S_SZ + j * CHLEN) * DINNER + e;
    float A = 0.f, Bv = -__builtin_inff();
    for (int s = 0; s < CHLEN; ++s) {
        __builtin_prefetch(LC + base + (size_t)32 * DINNER, 0, 3);
        __builtin_prefetch(LV + base + (size_t)32 * DINNER, 0, 3);
        float lc = LC[base], lv = LV[base];
        A += lc;
        Bv = logaddexpf_(Bv + lc, lv);
        base += DINNER;
    }
    CA[tid] = A;
    CB[tid] = Bv;
}

// ---------- scan phase B: serial combine over 8 chunks -> carry-in per chunk ----------
__global__ void scan_combine_kernel(const float* __restrict__ CA,
                                    const float* __restrict__ CB,
                                    float* __restrict__ LIN) {
    int tid = blockIdx.x * blockDim.x + threadIdx.x;   // 8192 chains
    int e = tid & (DINNER - 1);
    int b = tid >> 11;
    float l = -__builtin_inff();
#pragma unroll
    for (int j = 0; j < NCHUNK; ++j) {
        int idx = ((b << 3) + j) * DINNER + e;
        LIN[idx] = l;                                   // carry-in for chunk j
        l = logaddexpf_(CA[idx] + l, CB[idx]);
    }
}

// ---------- scan phase C: apply with carry, emit h = exp(l) as bf16 ----------
__global__ void scan_apply_kernel(const float* __restrict__ LC,
                                  const float* __restrict__ LV,
                                  const float* __restrict__ LIN,
                                  unsigned short* __restrict__ H) {
    int tid = blockIdx.x * blockDim.x + threadIdx.x;
    int e  = tid & (DINNER - 1);
    int bj = tid >> 11;
    int b = bj >> 3, j = bj & 7;
    size_t base = ((size_t)b * S_SZ + j * CHLEN) * DINNER + e;
    float l = LIN[tid];
    for (int s = 0; s < CHLEN; ++s) {
        __builtin_prefetch(LC + base + (size_t)32 * DINNER, 0, 3);
        __builtin_prefetch(LV + base + (size_t)32 * DINNER, 0, 3);
        l = logaddexpf_(l + LC[base], LV[base]);
        H[base] = f32_to_bf16(__expf(l));
        base += DINNER;
    }
}

// ---------- GEMM2: out = H @ W_out (f32 out), same pipelined structure ----------
__global__ __launch_bounds__(256) void gemm2_kernel(
    const unsigned short* __restrict__ Hb,    // [16384][2048] bf16
    const unsigned short* __restrict__ Wob,   // [2048][1024]  bf16
    float* __restrict__ Out) {                // [16384][1024] f32
    __shared__ uint4 smemq[1536];                        // 24 KB
    unsigned short* SA = reinterpret_cast<unsigned short*>(smemq);  // 2 x 128x32
    unsigned short* SB = SA + 2 * 4096;                             // 2 x 64x32 (T)

    const int t    = threadIdx.x;
    const int lane = t & 31;
    const int wave = t >> 5;
    const int wm   = wave >> 1;
    const int wn   = wave & 1;
    const int rowBase = blockIdx.y * 128;
    const int n0      = blockIdx.x * 64;

    v8f acc[2][2];
#pragma unroll
    for (int i = 0; i < 2; ++i)
#pragma unroll
        for (int j = 0; j < 2; ++j) acc[i][j] = zero8();

    const int arow = wm * 32 + (lane & 15);
    const int kbA  = (lane >> 4) * 8;
    const int bcol = wn * 32 + (lane & 15);
    const int kbB  = (lane >> 4) * 8;

    const int kkB = t >> 3;            // 0..31
    const int cB  = t & 7;             // 8 cols each

    union BStage { uint4 q; unsigned short s[8]; };

    auto issueA = [&](int kt, int buf) {
        const int k0 = kt * 32;
        unsigned short* dst = SA + buf * 4096;
#pragma unroll
        for (int it = 0; it < 2; ++it) {
            int lin = t + it * 256;
            int r = lin >> 2, c = lin & 3;
            const unsigned short* src = Hb + (size_t)(rowBase + r) * DINNER + k0 + c * 8;
            unsigned loff = (unsigned)(uintptr_t)(dst + r * 32 + c * 8);
            async_load_b128(loff, src);
        }
    };
    auto loadBregs = [&](int kt, BStage& u) {
        const unsigned short* src = Wob + (size_t)(kt * 32 + kkB) * DIM_ + n0 + cB * 8;
        u.q = *reinterpret_cast<const uint4*>(src);
    };
    auto storeB = [&](int buf, const BStage& u) {
        unsigned short* dst = SB + buf * 2048;
#pragma unroll
        for (int i = 0; i < 8; ++i) dst[(cB * 8 + i) * 32 + kkB] = u.s[i];
    };

    BStage bs;
    issueA(0, 0);
    loadBregs(0, bs);

    for (int kt = 0; kt < DINNER / 32; ++kt) {
        const int buf = kt & 1;
        storeB(buf, bs);
        wait_asynccnt0();
        __syncthreads();
        if (kt + 1 < DINNER / 32) {
            issueA(kt + 1, buf ^ 1);
            loadBregs(kt + 1, bs);
        }
        const unsigned short* As  = SA + buf * 4096;
        const unsigned short* Bst = SB + buf * 2048;

        v16bf a0 = load_fragA(As, arow,      kbA);
        v16bf a1 = load_fragA(As, arow + 16, kbA);
        v16bf b0 = load_fragB(Bst, bcol,      kbB);
        v16bf b1 = load_fragB(Bst, bcol + 16, kbB);

        acc[0][0] = __builtin_amdgcn_wmma_f32_16x16x32_bf16(false, a0, false, b0, (short)0, acc[0][0], false, false);
        acc[0][1] = __builtin_amdgcn_wmma_f32_16x16x32_bf16(false, a0, false, b1, (short)0, acc[0][1], false, false);
        acc[1][0] = __builtin_amdgcn_wmma_f32_16x16x32_bf16(false, a1, false, b0, (short)0, acc[1][0], false, false);
        acc[1][1] = __builtin_amdgcn_wmma_f32_16x16x32_bf16(false, a1, false, b1, (short)0, acc[1][1], false, false);
    }

#pragma unroll
    for (int fm = 0; fm < 2; ++fm)
#pragma unroll
        for (int fn = 0; fn < 2; ++fn)
#pragma unroll
            for (int i = 0; i < 8; ++i) {
                int row = rowBase + wm * 32 + fm * 16 + i + ((lane >> 4) << 3);
                int col = n0 + wn * 32 + fn * 16 + (lane & 15);
                Out[(size_t)row * DIM_ + col] = acc[fm][fn][i];
            }
}

// ---------- launch ----------
extern "C" void kernel_launch(void* const* d_in, const int* in_sizes, int n_in,
                              void* d_out, int out_size, void* d_ws, size_t ws_size,
                              hipStream_t stream) {
    (void)in_sizes; (void)n_in; (void)out_size; (void)ws_size;
    const float* x    = (const float*)d_in[0];   // [4,4096,1024]
    const float* Whg  = (const float*)d_in[1];   // [1024,4096]
    const float* Wout = (const float*)d_in[2];   // [2048,1024]
    float* out = (float*)d_out;

    char* ws = (char*)d_ws;
    size_t off = 0;
    unsigned short* Xb    = (unsigned short*)(ws + off); off += (size_t)M_ALL * DIM_ * 2;      // 32 MB
    unsigned short* Whgb  = (unsigned short*)(ws + off); off += (size_t)DIM_ * 2 * DINNER * 2; // 8 MB
    unsigned short* Woutb = (unsigned short*)(ws + off); off += (size_t)DINNER * DIM_ * 2;     // 4 MB
    float* LC = (float*)(ws + off); off += (size_t)M_ALL * DINNER * 4;                         // 128 MB
    float* LV = (float*)(ws + off); off += (size_t)M_ALL * DINNER * 4;                         // 128 MB
    unsigned short* H = (unsigned short*)(ws + off); off += (size_t)M_ALL * DINNER * 2;        // 64 MB
    float* CA  = (float*)(ws + off); off += (size_t)B_SZ * NCHUNK * DINNER * 4;
    float* CB  = (float*)(ws + off); off += (size_t)B_SZ * NCHUNK * DINNER * 4;
    float* LIN = (float*)(ws + off); off += (size_t)B_SZ * NCHUNK * DINNER * 4;

    // f32 -> bf16 once per call (deterministic)
    cvt_bf16_kernel<<<4096, 256, 0, stream>>>(x,    Xb,    M_ALL * DIM_);
    cvt_bf16_kernel<<<2048, 256, 0, stream>>>(Whg,  Whgb,  DIM_ * 2 * DINNER);
    cvt_bf16_kernel<<<1024, 256, 0, stream>>>(Wout, Woutb, DINNER * DIM_);

    // GEMM1 fused with split + softplus/log-g activation
    gemm1_act_kernel<<<dim3(DINNER / 64, M_ALL / 128), 256, 0, stream>>>(Xb, Whgb, LC, LV);

    // chunked log-space linear-recurrence scan over sequence axis
    scan_reduce_kernel <<<(B_SZ * NCHUNK * DINNER) / 256, 256, 0, stream>>>(LC, LV, CA, CB);
    scan_combine_kernel<<<(B_SZ * DINNER) / 256,          256, 0, stream>>>(CA, CB, LIN);
    scan_apply_kernel  <<<(B_SZ * NCHUNK * DINNER) / 256, 256, 0, stream>>>(LC, LV, LIN, H);

    // GEMM2: out = exp(log_h) @ W_out
    gemm2_kernel<<<dim3(DIM_ / 64, M_ALL / 128), 256, 0, stream>>>(H, Woutb, out);
}
